// GeometricAttention_30253749633442
// MI455X (gfx1250) — matrix operand order repaired and checked by reading, not compiled, not run on hardware
//
#include <hip/hip_runtime.h>
#include <cstdint>
#include <cstddef>

// ---------------------------------------------------------------------------
// GeometricAttention for MI455X (gfx1250, wave32, WMMA).
//
// scores = (q_feat . k_feat)/sqrt(192); softmax over keys; out = attn @ V.
// D_qk = 192, D_v = 288, N = 2048, BH = 16.  Compute-bound (~64 GFLOP vs
// ~150 MB) -> both GEMMs on v_wmma_f32_16x16x32_bf16 with f32 accumulate.
// Features precomputed to bf16 (scale & exp(log_w) folded into Qf); V stored
// transposed so every WMMA A/B operand loads row-major with the ISA layouts.
// Fragment loads are explicitly software-pipelined so the backend can issue
// partial s_wait_loadcnt instead of serial load->wait0->wmma ladders.
// ---------------------------------------------------------------------------

typedef __attribute__((ext_vector_type(16))) __bf16       bf16x16;
typedef __attribute__((ext_vector_type(8)))  float        f32x8;
typedef __attribute__((ext_vector_type(4)))  float        f32x4;
typedef __attribute__((ext_vector_type(4)))  unsigned int u32x4;

#define DEVI static __device__ __forceinline__

constexpr int   NB   = 2, NH = 8, NSEQ = 2048, CMV = 16, CS = 32;
constexpr int   BH   = NB * NH;
constexpr int   DQ   = 192;          // 128 ip + 32 norm + 32 raw
constexpr int   DV   = 288;          // 256 mv + 32 s
constexpr float SCALE = 0.07216878364870323f;   // 1/sqrt(192)
constexpr float EPS   = 0.001f;

union FragAB { u32x4 u[2]; bf16x16 f; };

// A-operand (16x32 bf16): lane l holds row (l&15); lanes 0-15: K=0..15,
// lanes 16-31: K=16..31 (contiguous 16/lane). `base` includes +hi*16 elements.
DEVI FragAB load_fragA(const __bf16* base) {
  FragAB r;
  r.u[0] = *reinterpret_cast<const u32x4*>(base);
  r.u[1] = *reinterpret_cast<const u32x4*>(base + 8);
  return r;
}
// B-operand (32x16 bf16): lane l holds col (l&15); lanes 0-15: K=0..15,
// lanes 16-31: K=16..31 (contiguous 16/lane). `base` includes +hi*16 elems.
DEVI FragAB load_fragB(const __bf16* base) {
  FragAB r;
  r.u[0] = *reinterpret_cast<const u32x4*>(base);
  r.u[1] = *reinterpret_cast<const u32x4*>(base + 8);
  return r;
}

DEVI unsigned pack_bf16(float x, float y) {
  union { __bf16 b[2]; unsigned u; } r;
  r.b[0] = (__bf16)x; r.b[1] = (__bf16)y;
  return r.u;
}

DEVI f32x8 wmma_bf16(bf16x16 a, bf16x16 b, f32x8 c) {
  return __builtin_amdgcn_wmma_f32_16x16x32_bf16(
      false, a, false, b, (short)0, c, false, false);
}

DEVI f32x8 vzero8() {
  f32x8 z;
#pragma unroll
  for (int i = 0; i < 8; ++i) z[i] = 0.f;
  return z;
}

// ---------------------------------------------------------------------------
// Prep 1: Qf/Kf bf16 feature rows (one wave32 per sequence row).
// ---------------------------------------------------------------------------
__global__ __launch_bounds__(256)
void prep_qk_kernel(const float* __restrict__ q_mv, const float* __restrict__ k_mv,
                    const float* __restrict__ q_s,  const float* __restrict__ k_s,
                    const float* __restrict__ lw,
                    __bf16* __restrict__ Qf, __bf16* __restrict__ Kf)
{
  const int row  = blockIdx.x * 8 + (threadIdx.x >> 5);   // (bh*N + n)
  const int lane = threadIdx.x & 31;
  const int h    = (row >> 11) & 7;                       // N = 2048
  const float* qm = q_mv + (size_t)row * 256;
  const float* km = k_mv + (size_t)row * 256;
  __bf16* qo = Qf + (size_t)row * DQ;
  __bf16* ko = Kf + (size_t)row * DQ;
#pragma unroll
  for (int i = 0; i < 4; ++i) {
    const int idx = lane * 4 + i;                         // 0..127
    const int c   = idx >> 3;
    const int e   = (int)((0xEA984320u >> ((idx & 7) * 4)) & 0xF); // _IP_IDX
    const float w = __expf(lw[h * 16 + c]);
    qo[idx] = (__bf16)(SCALE * w * qm[c * 16 + e]);
    ko[idx] = (__bf16)(km[c * 16 + e]);
  }
  const float qs = q_s[(size_t)row * 32 + lane];
  const float ks = k_s[(size_t)row * 32 + lane];
  qo[128 + lane] = (__bf16)(SCALE * qs / (qs * qs + EPS));
  qo[160 + lane] = (__bf16)(SCALE * qs);
  ko[128 + lane] = (__bf16)(ks / (ks * ks + EPS));
  ko[160 + lane] = (__bf16)ks;
}

// ---------------------------------------------------------------------------
// Prep 2: VfT[bh][d][n] = v_flat[bh][n][d]  (LDS 32x32 tile transpose).
// ---------------------------------------------------------------------------
__global__ __launch_bounds__(256)
void prep_v_kernel(const float* __restrict__ v_mv, const float* __restrict__ v_s,
                   __bf16* __restrict__ VfT)
{
  __shared__ float tile[32][33];
  const int bh = blockIdx.z;
  const int d0 = blockIdx.y * 32;
  const int n0 = blockIdx.x * 32;
  const int tx = threadIdx.x, ty = threadIdx.y;
  const float* src; int rs;
  if (d0 < 256) { src = v_mv + (size_t)bh * NSEQ * 256 + d0;         rs = 256; }
  else          { src = v_s  + (size_t)bh * NSEQ * 32  + (d0 - 256); rs = 32;  }
#pragma unroll
  for (int i = 0; i < 32; i += 8)
    tile[ty + i][tx] = src[(size_t)(n0 + ty + i) * rs + tx];
  __syncthreads();
  __bf16* dst = VfT + ((size_t)bh * DV + d0) * (size_t)NSEQ + n0;
#pragma unroll
  for (int i = 0; i < 32; i += 8)
    dst[(size_t)(ty + i) * NSEQ + tx] = (__bf16)tile[tx][ty + i];
}

// ---------------------------------------------------------------------------
// Flash attention: block = 256 (8 waves), wave owns 16 queries.
// Per 32-key step: S^T = Kf.Qf^T (12 wmma), online softmax (per-lane state,
// xor-16 shuffles), O^T += V^T.P^T (18 wmma).  All fragment streams are
// software-pipelined (QK depth 1, PV depth 2).
// ---------------------------------------------------------------------------
__global__ __launch_bounds__(256, 1)
void attn_kernel(const __bf16* __restrict__ Qf, const __bf16* __restrict__ Kf,
                 const __bf16* __restrict__ VfT,
                 float* __restrict__ out_mv, float* __restrict__ out_s)
{
  const int bh   = blockIdx.y;
  const int wave = threadIdx.x >> 5;
  const int lane = threadIdx.x & 31;
  const int lr   = lane & 15;
  const int hi   = lane >> 4;
  const int q0   = blockIdx.x * 128 + wave * 16;

  // Resident Q B-fragments: this lane's query column, 6 chunks of K=32.
  bf16x16 qf[6];
  {
    const __bf16* qrow = Qf + ((size_t)bh * NSEQ + q0 + lr) * DQ + hi * 16;
#pragma unroll
    for (int c = 0; c < 6; ++c) qf[c] = load_fragB(qrow + c * 32).f;
  }

  f32x8 acc[18];                       // O^T: 288 v-dims x 16 queries
#pragma unroll
  for (int j = 0; j < 18; ++j) acc[j] = vzero8();
  float m = -__builtin_inff(), lsum = 0.f;

  const __bf16* kbase = Kf  + ((size_t)bh * NSEQ) * DQ + (size_t)lr * DQ + hi * 16;
  const __bf16* vbase = VfT + ((size_t)bh * DV) * (size_t)NSEQ + (size_t)lr * NSEQ + hi * 16;

  for (int t = 0; t < NSEQ; t += 32) {
    const __bf16* k0 = kbase + (size_t)t * DQ;       // keys t..t+15
    const __bf16* k1 = k0 + 16 * DQ;                 // keys t+16..t+31

    // --- S^T = Kf . Qf^T, depth-1 pipelined A-fragment stream ---
    f32x8 s0 = vzero8(), s1 = vzero8();
    FragAB ka = load_fragA(k0), kb = load_fragA(k1);
#pragma unroll
    for (int c = 0; c < 6; ++c) {
      FragAB na, nb;
      if (c + 1 < 6) {
        na = load_fragA(k0 + (c + 1) * 32);
        nb = load_fragA(k1 + (c + 1) * 32);
      }
      s0 = wmma_bf16(ka.f, qf[c], s0);
      s1 = wmma_bf16(kb.f, qf[c], s1);
      if (c + 1 < 6) { ka = na; kb = nb; }
    }
    if (t + 32 < NSEQ) {                             // gfx1250 global_prefetch_b8
      __builtin_prefetch(k0 + 32 * DQ, 0, 3);
      __builtin_prefetch(vbase + t + 32, 0, 3);
    }

    // --- online softmax (1/sqrt(192) already folded into Qf) ---
    float mt = s0[0];
#pragma unroll
    for (int v = 1; v < 8; ++v) mt = fmaxf(mt, s0[v]);
#pragma unroll
    for (int v = 0; v < 8; ++v) mt = fmaxf(mt, s1[v]);
    mt = fmaxf(mt, __shfl_xor(mt, 16, 32));          // combine lane halves
    const float mnew = fmaxf(m, mt);

    // exp -> rowsum -> bf16 pack fused per pair (short live ranges).
    float rsum = 0.f;
    unsigned pa[4], pb[4];
#pragma unroll
    for (int j = 0; j < 4; ++j) {
      const float e0 = __expf(s0[2 * j] - mnew);
      const float e1 = __expf(s0[2 * j + 1] - mnew);
      const float f0 = __expf(s1[2 * j] - mnew);
      const float f1 = __expf(s1[2 * j + 1] - mnew);
      rsum += (e0 + e1) + (f0 + f1);
      pa[j] = pack_bf16(e0, e1);
      pb[j] = pack_bf16(f0, f1);
    }
    rsum += __shfl_xor(rsum, 16, 32);
    const float corr = __expf(m - mnew);
    lsum = lsum * corr + rsum;
    m = mnew;

    // --- pack P^T into B-fragment (K=32 keys, N=16 queries) ---
    // C-layout halves hold keys {0..7|8..15} (s0) / {16..23|24..31} (s1);
    // contiguous B order (matching V^T A loads): keys 0..15 on lanes 0-15, 16..31 on lanes 16-31.
    FragAB pf;
#pragma unroll
    for (int j = 0; j < 4; ++j) {
      const unsigned xa = __shfl_xor(pa[j], 16, 32);
      const unsigned xb = __shfl_xor(pb[j], 16, 32);
      pf.u[0][j] = hi ? xb : pa[j];
      pf.u[1][j] = hi ? pb[j] : xa;
    }

    // --- O^T += V^T . P^T, depth-2 pipelined A-fragment stream.
    // acc rescale interleaved so VALU muls co-execute with XDL WMMAs.
    const __bf16* vp = vbase + t;
    FragAB v0 = load_fragA(vp);
    FragAB v1 = load_fragA(vp + (size_t)16 * NSEQ);
#pragma unroll
    for (int vt = 0; vt < 18; ++vt) {
      FragAB nx;
      if (vt + 2 < 18) nx = load_fragA(vp + (size_t)((vt + 2) * 16) * NSEQ);
      acc[vt] *= corr;
      acc[vt] = wmma_bf16((vt & 1) ? v1.f : v0.f, pf.f, acc[vt]);
      if (vt + 2 < 18) { if (vt & 1) v1 = nx; else v0 = nx; }
    }
  }

  // --- epilogue: O = O^T / l, scatter back row-major ---
  const float inv = 1.f / lsum;
  const int   q   = q0 + lr;
  float* om = out_mv + ((size_t)bh * NSEQ + q) * 256 + hi * 8;
#pragma unroll
  for (int vt = 0; vt < 16; ++vt) {
    f32x4 w0 = { acc[vt][0] * inv, acc[vt][1] * inv, acc[vt][2] * inv, acc[vt][3] * inv };
    f32x4 w1 = { acc[vt][4] * inv, acc[vt][5] * inv, acc[vt][6] * inv, acc[vt][7] * inv };
    *reinterpret_cast<f32x4*>(om + vt * 16)     = w0;
    *reinterpret_cast<f32x4*>(om + vt * 16 + 4) = w1;
  }
  float* os = out_s + ((size_t)bh * NSEQ + q) * 32 + hi * 8;
#pragma unroll
  for (int vt = 16; vt < 18; ++vt) {
    f32x4 w0 = { acc[vt][0] * inv, acc[vt][1] * inv, acc[vt][2] * inv, acc[vt][3] * inv };
    f32x4 w1 = { acc[vt][4] * inv, acc[vt][5] * inv, acc[vt][6] * inv, acc[vt][7] * inv };
    *reinterpret_cast<f32x4*>(os + (vt - 16) * 16)     = w0;
    *reinterpret_cast<f32x4*>(os + (vt - 16) * 16 + 4) = w1;
  }
}

// ---------------------------------------------------------------------------
extern "C" void kernel_launch(void* const* d_in, const int* in_sizes, int n_in,
                              void* d_out, int out_size, void* d_ws, size_t ws_size,
                              hipStream_t stream)
{
  (void)in_sizes; (void)n_in; (void)out_size; (void)ws_size;
  const float* q_mv = (const float*)d_in[0];
  const float* k_mv = (const float*)d_in[1];
  const float* v_mv = (const float*)d_in[2];
  const float* q_s  = (const float*)d_in[3];
  const float* k_s  = (const float*)d_in[4];
  const float* v_s  = (const float*)d_in[5];
  const float* lw   = (const float*)d_in[6];

  float* out_mv = (float*)d_out;
  float* out_s  = out_mv + (size_t)BH * NSEQ * 256;

  // Workspace: Qf (12.6 MB) | Kf (12.6 MB) | VfT (18.9 MB), all bf16.
  __bf16* Qf  = (__bf16*)d_ws;
  __bf16* Kf  = Qf + (size_t)BH * NSEQ * DQ;
  __bf16* VfT = Kf + (size_t)BH * NSEQ * DQ;

  prep_qk_kernel<<<dim3(BH * NSEQ / 8), dim3(256), 0, stream>>>(
      q_mv, k_mv, q_s, k_s, lw, Qf, Kf);
  prep_v_kernel<<<dim3(NSEQ / 32, DV / 32, BH), dim3(32, 8), 0, stream>>>(
      v_mv, v_s, VfT);
  attn_kernel<<<dim3(NSEQ / 128, BH), dim3(256), 0, stream>>>(
      Qf, Kf, VfT, out_mv, out_s);
}
